// ImprovedGNN_3461743640929
// MI455X (gfx1250) — matrix operand order repaired and checked by reading, not compiled, not run on hardware
//
#include <hip/hip_runtime.h>
#include <hip/hip_bf16.h>
#include <math.h>

typedef __bf16 bf16_t;
typedef __attribute__((ext_vector_type(16))) __bf16 v16bf;
typedef __attribute__((ext_vector_type(8)))  float  v8f;

#define EPS 1e-5f

// ---------------------------------------------------------------------------
// Small utility kernels
// ---------------------------------------------------------------------------
__global__ void k_fill_f32(float* p, float v, long long n) {
    long long i = (long long)blockIdx.x * blockDim.x + threadIdx.x;
    long long stride = (long long)gridDim.x * blockDim.x;
    for (; i < n; i += stride) p[i] = v;
}

__global__ void k_fill_u32(unsigned* p, unsigned v, long long n) {
    long long i = (long long)blockIdx.x * blockDim.x + threadIdx.x;
    long long stride = (long long)gridDim.x * blockDim.x;
    for (; i < n; i += stride) p[i] = v;
}

__global__ void k_f32_to_bf16(const float* __restrict__ src, bf16_t* __restrict__ dst, long long n) {
    long long i = (long long)blockIdx.x * blockDim.x + threadIdx.x;
    long long stride = (long long)gridDim.x * blockDim.x;
    for (; i < n; i += stride) dst[i] = (bf16_t)src[i];
}

// src is R x C f32 ; dst is Rp x Cp bf16, zero-padded (row-major)
__global__ void k_pad_convert(const float* __restrict__ src, bf16_t* __restrict__ dst,
                              int R, int C, int Rp, int Cp) {
    long long n = (long long)Rp * Cp;
    long long i = (long long)blockIdx.x * blockDim.x + threadIdx.x;
    long long stride = (long long)gridDim.x * blockDim.x;
    for (; i < n; i += stride) {
        int r = (int)(i / Cp), c = (int)(i % Cp);
        float v = (r < R && c < C) ? src[(long long)r * C + c] : 0.0f;
        dst[i] = (bf16_t)v;
    }
}

// Repack weight K x N (f32, row-major) into WMMA B-operand layout (bf16):
// per (32-K x 16-N) tile: 512 contiguous bf16 = 32 lanes x 16 elems.
// lane = half*16 + l -> col = nt*16 + l ; elem j -> krow = kt*32 + half*16 + j
// Kp is K zero-padded up to a multiple of 32.
__global__ void k_pack_w(const float* __restrict__ src, bf16_t* __restrict__ dst,
                         int K, int N, int Kp) {
    long long total = (long long)Kp * N;
    long long i = (long long)blockIdx.x * blockDim.x + threadIdx.x;
    long long stride = (long long)gridDim.x * blockDim.x;
    int ntiles = N >> 4;
    for (; i < total; i += stride) {
        int blk    = (int)(i >> 9);       // which 512-elem tile
        int within = (int)(i & 511);
        int lane = within >> 4, j = within & 15;
        int kt = blk / ntiles, nt = blk % ntiles;
        int half = lane >> 4, l = lane & 15;
        int k   = kt * 32 + half * 16 + j;
        int col = nt * 16 + l;
        float v = (k < K) ? src[(size_t)k * N + col] : 0.0f;
        dst[i] = (bf16_t)v;
    }
}

// inference BN folded into scale/shift: y = x*scale + shift
__global__ void k_bn_fuse(const float* g, const float* b, const float* m, const float* v,
                          float* scale, float* shift, int dim) {
    int i = blockIdx.x * blockDim.x + threadIdx.x;
    if (i < dim) {
        float s = g[i] * rsqrtf(v[i] + EPS);
        scale[i] = s;
        shift[i] = b[i] - m[i] * s;
    }
}

// ---------------------------------------------------------------------------
// Graph structure kernels
// ---------------------------------------------------------------------------
__global__ void k_edge_deg(const long long* __restrict__ ei, float* deg, long long E) {
    long long i = (long long)blockIdx.x * blockDim.x + threadIdx.x;
    long long stride = (long long)gridDim.x * blockDim.x;
    for (; i < E; i += stride) atomicAdd(&deg[(int)ei[E + i]], 1.0f);
}

__global__ void k_node_norm(const float* __restrict__ deg, float* dinv, float* invd, int N) {
    int i = blockIdx.x * blockDim.x + threadIdx.x;
    if (i < N) {
        float d = deg[i] + 1.0f;           // self-loop
        dinv[i] = rsqrtf(d);
        invd[i] = 1.0f / d;
    }
}

// agg[dst] += t[src] * dinv[src]*dinv[dst]  ; one block per edge, F threads
__global__ void k_edge_scatter(const long long* __restrict__ ei, const float* __restrict__ t,
                               const float* __restrict__ dinv, float* agg, long long E, int F) {
    long long e = blockIdx.x;
    int f = threadIdx.x;
    int s = (int)ei[e];
    int d = (int)ei[E + e];
    float norm = dinv[s] * dinv[d];
    atomicAdd(&agg[(size_t)d * F + f], t[(size_t)s * F + f] * norm);
}

// out = relu( (agg + t*invdeg + bias)*bnscale + bnshift [+ res] )
__global__ void k_gcn_finalize(const float* __restrict__ agg, const float* __restrict__ t,
                               const float* __restrict__ invd, const float* __restrict__ bias,
                               const float* __restrict__ bns, const float* __restrict__ bnh,
                               const float* __restrict__ res, float* out, long long N, int F) {
    long long n_el = N * F;
    long long i = (long long)blockIdx.x * blockDim.x + threadIdx.x;
    long long stride = (long long)gridDim.x * blockDim.x;
    for (; i < n_el; i += stride) {
        int f = (int)(i % F);
        long long n = i / F;
        float v = (agg[i] + t[i] * invd[n] + bias[f]) * bns[f] + bnh[f];
        if (res) v += res[i];
        out[i] = fmaxf(v, 0.0f);
    }
}

// ---------------------------------------------------------------------------
// WMMA bf16 GEMM: out[M x N] = act( (A[M x K] @ W[K x N] + bias)*bnscale + bnshift )
// A: bf16 row-major (K multiple of 32). W: packed B-operand layout (k_pack_w).
// Wave -> 16 rows x 64 cols (4 accumulators); block = 4 waves = 64 rows x 64 cols.
// Requires N % 64 == 0, M % 16 == 0.
// ---------------------------------------------------------------------------
__global__ void k_gemm_wmma(const bf16_t* __restrict__ A, const bf16_t* __restrict__ Wp,
                            const float* __restrict__ bias, const float* __restrict__ bns,
                            const float* __restrict__ bnh, float* __restrict__ out,
                            int M, int K, int N, int relu) {
    const int lane = threadIdx.x & 31;
    const int wave = threadIdx.x >> 5;
    const int row0 = blockIdx.y * 64 + wave * 16;
    const int col0 = blockIdx.x * 64;
    if (row0 >= M) return;                       // wave-uniform (M % 16 == 0)

    const int half = lane >> 4;                  // 0: lanes 0-15, 1: lanes 16-31
    const int l    = lane & 15;
    const int ntiles = N >> 4;

    v8f acc0 = {}, acc1 = {}, acc2 = {}, acc3 = {};

    const bf16_t* arow  = A + (size_t)(row0 + l) * K;
    const bf16_t* wbase = Wp + (size_t)(col0 >> 4) * 512 + (size_t)lane * 16;

    for (int k0 = 0; k0 < K; k0 += 32) {
        // A fragment 16x32: half 0 -> K {0..15}, half 1 -> K {16..31}
        v16bf a;
        const int kb = k0 + half * 16;
#pragma unroll
        for (int i = 0; i < 8; i++) a[i] = arow[kb + i];
#pragma unroll
        for (int i = 0; i < 8; i++) a[8 + i] = arow[kb + 8 + i];

        // B fragments: one aligned 32-byte vector load per tile (packed layout)
        const bf16_t* wk = wbase + (size_t)(k0 >> 5) * ((size_t)ntiles * 512);
        v16bf b0 = *(const v16bf*)(wk);
        v16bf b1 = *(const v16bf*)(wk + 512);
        v16bf b2 = *(const v16bf*)(wk + 1024);
        v16bf b3 = *(const v16bf*)(wk + 1536);

        acc0 = __builtin_amdgcn_wmma_f32_16x16x32_bf16(false, a, false, b0, (short)0, acc0, false, false);
        acc1 = __builtin_amdgcn_wmma_f32_16x16x32_bf16(false, a, false, b1, (short)0, acc1, false, false);
        acc2 = __builtin_amdgcn_wmma_f32_16x16x32_bf16(false, a, false, b2, (short)0, acc2, false, false);
        acc3 = __builtin_amdgcn_wmma_f32_16x16x32_bf16(false, a, false, b3, (short)0, acc3, false, false);
    }

    // C/D layout: lane l, half h -> col ; VGPR r -> row = row0 + r + 8*h
    auto store_tile = [&](const v8f& acc, int tc0) {
        const int col = tc0 + l;
        const float bi = bias ? bias[col] : 0.0f;
        const float bs = bns ? bns[col] : 1.0f;
        const float bh = bnh ? bnh[col] : 0.0f;
#pragma unroll
        for (int r = 0; r < 8; r++) {
            int row = row0 + r + 8 * half;
            float v = (acc[r] + bi) * bs + bh;
            if (relu) v = fmaxf(v, 0.0f);
            out[(size_t)row * N + col] = v;
        }
    };
    store_tile(acc0, col0);
    store_tile(acc1, col0 + 16);
    store_tile(acc2, col0 + 32);
    store_tile(acc3, col0 + 48);
}

// ---------------------------------------------------------------------------
// Pooling
// ---------------------------------------------------------------------------
__global__ void k_pool_scatter(const float* __restrict__ h, const long long* __restrict__ batch,
                               float* addbuf, unsigned* maxbuf, float* counts, int F) {
    int n = blockIdx.x;
    int f = threadIdx.x;
    int g = (int)batch[n];
    float v = h[(size_t)n * F + f];
    atomicAdd(&addbuf[(size_t)g * F + f], v);
    atomicMax(&maxbuf[(size_t)g * F + f], __float_as_uint(v));  // v >= 0 post-relu
    if (f == 0) atomicAdd(&counts[g], 1.0f);
}

// pooled (G x 3F) = [mean | max | add], converted to bf16
__global__ void k_pool_build(const float* __restrict__ addbuf, const unsigned* __restrict__ maxbuf,
                             const float* __restrict__ counts, bf16_t* pooledb, int G, int F) {
    int idx = blockIdx.x * blockDim.x + threadIdx.x;
    if (idx >= G * F) return;
    int g = idx / F, f = idx % F;
    float c   = fmaxf(counts[g], 1.0f);
    float add = addbuf[idx];
    float mx  = __uint_as_float(maxbuf[idx]);
    int P = 3 * F;
    pooledb[(size_t)g * P + f]         = (bf16_t)(add / c);
    pooledb[(size_t)g * P + F + f]     = (bf16_t)mx;
    pooledb[(size_t)g * P + 2 * F + f] = (bf16_t)add;
}

// final: out[g] = z3[g,:] . Wp4 + bp4
__global__ void k_final(const float* __restrict__ z3, const float* __restrict__ Wp4,
                        const float* __restrict__ bp4, float* out, int G, int K) {
    int g = blockIdx.x * blockDim.x + threadIdx.x;
    if (g >= G) return;
    float s = bp4[0];
    for (int k = 0; k < K; k++) s += z3[(size_t)g * K + k] * Wp4[k];
    out[g] = s;
}

// ---------------------------------------------------------------------------
// Host-side launch
// ---------------------------------------------------------------------------
static inline int cdiv(long long a, long long b) { return (int)((a + b - 1) / b); }

extern "C" void kernel_launch(void* const* d_in, const int* in_sizes, int n_in,
                              void* d_out, int out_size, void* d_ws, size_t ws_size,
                              hipStream_t stream) {
    const long long N = in_sizes[0] / 16;   // 100000
    const long long E = in_sizes[1] / 2;    // 1600000
    const int G = 64, H = 128, H2 = 256, KIN = 32, P = 3 * H2;

    const float*      x     = (const float*)d_in[0];
    const long long*  ei    = (const long long*)d_in[1];
    const long long*  batch = (const long long*)d_in[2];
    // d_in[3] = num_graphs (scalar, hardcoded 64)
    int a = 4;
    const float *W_in=(const float*)d_in[a++], *b_in=(const float*)d_in[a++];
    const float *gin=(const float*)d_in[a++], *bin=(const float*)d_in[a++], *min_=(const float*)d_in[a++], *vin=(const float*)d_in[a++];
    const float *W1=(const float*)d_in[a++], *b1=(const float*)d_in[a++];
    const float *g1=(const float*)d_in[a++], *bb1=(const float*)d_in[a++], *m1=(const float*)d_in[a++], *v1=(const float*)d_in[a++];
    const float *W2=(const float*)d_in[a++], *b2=(const float*)d_in[a++];
    const float *g2=(const float*)d_in[a++], *bb2=(const float*)d_in[a++], *m2=(const float*)d_in[a++], *v2=(const float*)d_in[a++];
    const float *W3=(const float*)d_in[a++], *b3=(const float*)d_in[a++];
    const float *g3=(const float*)d_in[a++], *bb3=(const float*)d_in[a++], *m3=(const float*)d_in[a++], *v3=(const float*)d_in[a++];
    const float *W4=(const float*)d_in[a++], *b4=(const float*)d_in[a++];
    const float *g4=(const float*)d_in[a++], *bb4=(const float*)d_in[a++], *m4=(const float*)d_in[a++], *v4=(const float*)d_in[a++];
    const float *W_res=(const float*)d_in[a++], *b_res=(const float*)d_in[a++];
    const float *Wp1=(const float*)d_in[a++], *bp1=(const float*)d_in[a++];
    const float *gp1=(const float*)d_in[a++], *bbp1=(const float*)d_in[a++], *mp1=(const float*)d_in[a++], *vp1=(const float*)d_in[a++];
    const float *Wp2=(const float*)d_in[a++], *bp2=(const float*)d_in[a++];
    const float *gp2=(const float*)d_in[a++], *bbp2=(const float*)d_in[a++], *mp2=(const float*)d_in[a++], *vp2=(const float*)d_in[a++];
    const float *Wp3=(const float*)d_in[a++], *bp3=(const float*)d_in[a++];
    const float *Wp4=(const float*)d_in[a++], *bp4=(const float*)d_in[a++];
    float* out = (float*)d_out;

    // ---- bump allocator over workspace ----
    char* ws = (char*)d_ws;
    size_t off = 0;
    auto alloc = [&](size_t bytes) -> void* {
        off = (off + 255) & ~(size_t)255;
        void* p = ws + off;
        off += bytes;
        return p;
    };
    float*  bufA = (float*) alloc((size_t)N * H2 * 4);   // activations (f32)
    float*  bufB = (float*) alloc((size_t)N * H2 * 4);   // gemm result t
    float*  bufC = (float*) alloc((size_t)N * H2 * 4);   // agg
    float*  bufR = (float*) alloc((size_t)N * H2 * 4);   // residual
    bf16_t* S    = (bf16_t*)alloc((size_t)N * H2 * 2);   // bf16 staging of activations
    bf16_t* Xb   = (bf16_t*)alloc((size_t)N * KIN * 2);  // padded bf16 x
    float*  deg  = (float*) alloc((size_t)N * 4);
    float*  dinv = (float*) alloc((size_t)N * 4);
    float*  invd = (float*) alloc((size_t)N * 4);
    bf16_t* Winb  = (bf16_t*)alloc((size_t)KIN * H * 2);
    bf16_t* W1b   = (bf16_t*)alloc((size_t)H * H * 2);
    bf16_t* W2b   = (bf16_t*)alloc((size_t)H * H2 * 2);
    bf16_t* W3b   = (bf16_t*)alloc((size_t)H2 * H2 * 2);
    bf16_t* W4b   = (bf16_t*)alloc((size_t)H2 * H2 * 2);
    bf16_t* Wresb = (bf16_t*)alloc((size_t)H * H2 * 2);
    bf16_t* Wp1b  = (bf16_t*)alloc((size_t)P * H2 * 2);
    bf16_t* Wp2b  = (bf16_t*)alloc((size_t)H2 * H * 2);
    bf16_t* Wp3b  = (bf16_t*)alloc((size_t)H * 64 * 2);
    float *sin_=(float*)alloc(H*4), *hin_=(float*)alloc(H*4);
    float *s1=(float*)alloc(H*4),  *h1=(float*)alloc(H*4);
    float *s2=(float*)alloc(H2*4), *h2=(float*)alloc(H2*4);
    float *s3=(float*)alloc(H2*4), *h3=(float*)alloc(H2*4);
    float *s4=(float*)alloc(H2*4), *h4=(float*)alloc(H2*4);
    float *sp1=(float*)alloc(H2*4),*hp1=(float*)alloc(H2*4);
    float *sp2=(float*)alloc(H*4), *hp2=(float*)alloc(H*4);
    float*    addb = (float*)   alloc((size_t)G * H2 * 4);
    unsigned* maxb = (unsigned*)alloc((size_t)G * H2 * 4);
    float*    cnts = (float*)   alloc((size_t)G * 4);
    bf16_t*   Pb   = (bf16_t*)  alloc((size_t)G * P * 2);
    float*    Z1   = (float*)   alloc((size_t)G * H2 * 4);
    bf16_t*   Z1b  = (bf16_t*)  alloc((size_t)G * H2 * 2);
    float*    Z2   = (float*)   alloc((size_t)G * H * 4);
    bf16_t*   Z2b  = (bf16_t*)  alloc((size_t)G * H * 2);
    float*    Z3   = (float*)   alloc((size_t)G * 64 * 4);
    (void)ws_size; (void)n_in; (void)out_size;

    // ---- fold BN params ----
    k_bn_fuse<<<1,128,0,stream>>>(gin,bin,min_,vin,sin_,hin_,H);
    k_bn_fuse<<<1,128,0,stream>>>(g1,bb1,m1,v1,s1,h1,H);
    k_bn_fuse<<<2,128,0,stream>>>(g2,bb2,m2,v2,s2,h2,H2);
    k_bn_fuse<<<2,128,0,stream>>>(g3,bb3,m3,v3,s3,h3,H2);
    k_bn_fuse<<<2,128,0,stream>>>(g4,bb4,m4,v4,s4,h4,H2);
    k_bn_fuse<<<2,128,0,stream>>>(gp1,bbp1,mp1,vp1,sp1,hp1,H2);
    k_bn_fuse<<<1,128,0,stream>>>(gp2,bbp2,mp2,vp2,sp2,hp2,H);

    // ---- weights -> packed WMMA B layout (bf16); W_in K 16->32 zero-pad ----
    k_pack_w<<<cdiv((long long)KIN*H,256),256,0,stream>>>(W_in, Winb, 16, H, KIN);
    k_pack_w<<<cdiv((long long)H*H,256),256,0,stream>>>(W1, W1b, H, H, H);
    k_pack_w<<<cdiv((long long)H*H2,256),256,0,stream>>>(W2, W2b, H, H2, H);
    k_pack_w<<<cdiv((long long)H2*H2,256),256,0,stream>>>(W3, W3b, H2, H2, H2);
    k_pack_w<<<cdiv((long long)H2*H2,256),256,0,stream>>>(W4, W4b, H2, H2, H2);
    k_pack_w<<<cdiv((long long)H*H2,256),256,0,stream>>>(W_res, Wresb, H, H2, H);
    k_pack_w<<<cdiv((long long)P*H2,256),256,0,stream>>>(Wp1, Wp1b, P, H2, P);
    k_pack_w<<<cdiv((long long)H2*H,256),256,0,stream>>>(Wp2, Wp2b, H2, H, H2);
    k_pack_w<<<cdiv((long long)H*64,256),256,0,stream>>>(Wp3, Wp3b, H, 64, H);

    // ---- x -> bf16 padded to K=32 ----
    k_pad_convert<<<4096,256,0,stream>>>(x, Xb, (int)N, 16, (int)N, KIN);

    // ---- degrees & norms ----
    k_fill_f32<<<2048,256,0,stream>>>(deg, 0.0f, N);
    k_edge_deg<<<4096,256,0,stream>>>(ei, deg, E);
    k_node_norm<<<cdiv(N,256),256,0,stream>>>(deg, dinv, invd, (int)N);

    const dim3 blk(128);
    auto ggrid = [&](long long M, int Nout) { return dim3((unsigned)cdiv(Nout,64), (unsigned)cdiv(M,64)); };

    // ---- input layer: h = relu(bn(x @ W_in + b_in)) -> bufA (N x 128) ----
    k_gemm_wmma<<<ggrid(N,H), blk, 0, stream>>>(Xb, Winb, b_in, sin_, hin_, bufA, (int)N, KIN, H, 1);

    // ---- stage h_input bf16; residual h_res = h_input @ W_res + b_res -> bufR ----
    k_f32_to_bf16<<<8192,256,0,stream>>>(bufA, S, N*H);
    k_gemm_wmma<<<ggrid(N,H2), blk, 0, stream>>>(S, Wresb, b_res, nullptr, nullptr, bufR, (int)N, H, H2, 0);

    // ---- GCN layer 1 (F=128) ----
    k_gemm_wmma<<<ggrid(N,H), blk, 0, stream>>>(S, W1b, nullptr, nullptr, nullptr, bufB, (int)N, H, H, 0);
    k_fill_f32<<<8192,256,0,stream>>>(bufC, 0.0f, N*H);
    k_edge_scatter<<<(unsigned)E, H, 0, stream>>>(ei, bufB, dinv, bufC, E, H);
    k_gcn_finalize<<<8192,256,0,stream>>>(bufC, bufB, invd, b1, s1, h1, nullptr, bufA, N, H);

    // ---- GCN layer 2 (128 -> 256, + residual) ----
    k_f32_to_bf16<<<8192,256,0,stream>>>(bufA, S, N*H);
    k_gemm_wmma<<<ggrid(N,H2), blk, 0, stream>>>(S, W2b, nullptr, nullptr, nullptr, bufB, (int)N, H, H2, 0);
    k_fill_f32<<<8192,256,0,stream>>>(bufC, 0.0f, N*H2);
    k_edge_scatter<<<(unsigned)E, H2, 0, stream>>>(ei, bufB, dinv, bufC, E, H2);
    k_gcn_finalize<<<8192,256,0,stream>>>(bufC, bufB, invd, b2, s2, h2, bufR, bufA, N, H2);

    // ---- GCN layer 3 (256 -> 256) ----
    k_f32_to_bf16<<<8192,256,0,stream>>>(bufA, S, N*H2);
    k_gemm_wmma<<<ggrid(N,H2), blk, 0, stream>>>(S, W3b, nullptr, nullptr, nullptr, bufB, (int)N, H2, H2, 0);
    k_fill_f32<<<8192,256,0,stream>>>(bufC, 0.0f, N*H2);
    k_edge_scatter<<<(unsigned)E, H2, 0, stream>>>(ei, bufB, dinv, bufC, E, H2);
    k_gcn_finalize<<<8192,256,0,stream>>>(bufC, bufB, invd, b3, s3, h3, nullptr, bufA, N, H2);

    // ---- GCN layer 4 (256 -> 256) ----
    k_f32_to_bf16<<<8192,256,0,stream>>>(bufA, S, N*H2);
    k_gemm_wmma<<<ggrid(N,H2), blk, 0, stream>>>(S, W4b, nullptr, nullptr, nullptr, bufB, (int)N, H2, H2, 0);
    k_fill_f32<<<8192,256,0,stream>>>(bufC, 0.0f, N*H2);
    k_edge_scatter<<<(unsigned)E, H2, 0, stream>>>(ei, bufB, dinv, bufC, E, H2);
    k_gcn_finalize<<<8192,256,0,stream>>>(bufC, bufB, invd, b4, s4, h4, nullptr, bufA, N, H2);

    // ---- pooling: mean | max | add over batch ----
    k_fill_f32<<<64,256,0,stream>>>(addb, 0.0f, (long long)G*H2);
    k_fill_u32<<<64,256,0,stream>>>(maxb, 0u, (long long)G*H2);
    k_fill_f32<<<1,64,0,stream>>>(cnts, 0.0f, G);
    k_pool_scatter<<<(unsigned)N, H2, 0, stream>>>(bufA, batch, addb, maxb, cnts, H2);
    k_pool_build<<<cdiv((long long)G*H2,256),256,0,stream>>>(addb, maxb, cnts, Pb, G, H2);

    // ---- MLP head (WMMA gemms on 64 rows) ----
    k_gemm_wmma<<<ggrid(G,H2), blk, 0, stream>>>(Pb, Wp1b, bp1, sp1, hp1, Z1, G, P, H2, 1);
    k_f32_to_bf16<<<64,256,0,stream>>>(Z1, Z1b, (long long)G*H2);
    k_gemm_wmma<<<ggrid(G,H), blk, 0, stream>>>(Z1b, Wp2b, bp2, sp2, hp2, Z2, G, H2, H, 1);
    k_f32_to_bf16<<<32,256,0,stream>>>(Z2, Z2b, (long long)G*H);
    k_gemm_wmma<<<ggrid(G,64), blk, 0, stream>>>(Z2b, Wp3b, bp3, nullptr, nullptr, Z3, G, H, 64, 1);
    k_final<<<1,64,0,stream>>>(Z3, Wp4, bp4, out, G, 64);
}
